// VQ_VAE_Model_5970004541984
// MI455X (gfx1250) — matrix operand order repaired and checked by reading, not compiled, not run on hardware
//
#include <hip/hip_runtime.h>
#include <hip/hip_bf16.h>

typedef __attribute__((ext_vector_type(16))) _Float16 v16h;
typedef __attribute__((ext_vector_type(8)))  _Float16 v8h;
typedef __attribute__((ext_vector_type(8)))  float    v8f;

// ---------------------------------------------------------------------------
// Implicit-GEMM conv / transposed-conv on f16 NHWC activations.
// One wave per 16x(16*NB) tile of the (M = B*Ho*Wo) x (N = Cout) GEMM,
// K reordered as (ky, kx, ci).  Activations are [n][y][x][CINP] f16 with
// CINP % 32 == 0 (zero-padded), so each A fragment is two contiguous b128
// loads; B fragments are one contiguous 32B load from pre-packed f16 weights
// [CoutP][ky*kx][CINP].  One A fragment feeds NB=8 independent WMMA chains.
// A and B fragments both use contiguous k = 16*(lane/16) + s: no LDS/barriers.
// ACT: 0=none 1=relu 2=sigmoid.  OUTF32: write f32 NCHW (final layer).
// ---------------------------------------------------------------------------
template<int CINP, int HIN, int WIN, int COUT, int COUTP, int HOUT, int WOUT,
         int KH, int KW, int S, int P, int TRANS, int RELU_IN, int ACT,
         int OUTF32, int NB>
__global__ __launch_bounds__(32)
void conv_wmma(const _Float16* __restrict__ in, const _Float16* __restrict__ wp,
               const float* __restrict__ bias, const _Float16* __restrict__ res,
               void* __restrict__ outv)
{
    constexpr int KP = KH * KW * CINP;               // packed K per out-channel

    const int lane = threadIdx.x;
    const int m0   = blockIdx.x * 16;
    const int n0   = blockIdx.y * (16 * NB);

    // A fragment geometry: lane holds output pixel M = m0 + (lane&15)
    const int mA   = m0 + (lane & 15);
    const int hiA  = lane >> 4;                      // K-halfline select
    const int ox   = mA % WOUT;                      // powers of two -> shifts
    const int oy   = (mA / WOUT) % HOUT;
    const int nImg = mA / (WOUT * HOUT);
    const int iy0  = TRANS ? (oy + P) : (oy * S - P);
    const int ix0  = TRANS ? (ox + P) : (ox * S - P);
    const _Float16* __restrict__ inN = in + (long)nImg * HIN * WIN * CINP;

    // B fragment geometry: lane holds column N = (lane&15) of each tile
    const int nB   = lane & 15;
    const int kloB = (lane >> 4) * 16;

    v8f acc[NB];
#pragma unroll
    for (int t = 0; t < NB; ++t) acc[t] = (v8f){};

#pragma unroll
    for (int ky = 0; ky < KH; ++ky) {
#pragma unroll
        for (int kx = 0; kx < KW; ++kx) {
            // ---- tap-uniform input coordinate + validity ----
            int iy, ix;
            bool ok;
            if (TRANS) {
                const int ty = iy0 - ky;
                const int tx = ix0 - kx;
                ok = (ty >= 0) && (tx >= 0) &&
                     ((S == 1) || (((ty & (S - 1)) | (tx & (S - 1))) == 0));
                iy = ty / S;
                ix = tx / S;
            } else {
                iy = iy0 + ky;
                ix = ix0 + kx;
                ok = true;
            }
            ok = ok && (iy >= 0) && (iy < HIN) && (ix >= 0) && (ix < WIN);
            const _Float16* __restrict__ aptr = inN + (long)(iy * WIN + ix) * CINP;
            const int kofs = ky * KW + kx;

#pragma unroll
            for (int cb = 0; cb < CINP; cb += 32) {
                // ---- A fragment: two contiguous b128 loads (f16 NHWC) ----
                v8h lo = {}, hi = {};
                if (ok) {
                    lo = *(const v8h*)(aptr + cb + hiA * 16);
                    hi = *(const v8h*)(aptr + cb + 8 + hiA * 16);
                    if (RELU_IN) {
#pragma unroll
                        for (int j = 0; j < 8; ++j) {
                            lo[j] = lo[j] > (_Float16)0 ? lo[j] : (_Float16)0;
                            hi[j] = hi[j] > (_Float16)0 ? hi[j] : (_Float16)0;
                        }
                    }
                }
                const v16h a = __builtin_shufflevector(
                    lo, hi, 0, 1, 2, 3, 4, 5, 6, 7, 8, 9, 10, 11, 12, 13, 14, 15);

                // ---- NB x (contiguous packed-f16 B fragment, WMMA) ----
                const int kb = kofs * CINP + cb + kloB;
#pragma unroll
                for (int t = 0; t < NB; ++t) {
                    const int nCh = n0 + t * 16 + nB;
                    const v16h b = *(const v16h*)(wp + (long)nCh * KP + kb);
                    acc[t] = __builtin_amdgcn_wmma_f32_16x16x32_f16(
                                 false, a, false, b, (short)0, acc[t], false, false);
                }
            }
        }
    }

    // ------------------------- epilogue --------------------------------
#pragma unroll
    for (int t = 0; t < NB; ++t) {
        const int nCh = n0 + t * 16 + nB;
        const float bv = (COUT == COUTP || nCh < COUT) ? bias[nCh] : 0.0f;
#pragma unroll
        for (int r = 0; r < 8; ++r) {
            const int m  = m0 + r + (hiA << 3);      // C layout: M=r (+8 hi half)
            float v = acc[t][r] + bv;
            if (ACT == 1) v = fmaxf(v, 0.0f);
            else if (ACT == 2) v = 1.0f / (1.0f + __expf(-v));
            const int px = m % WOUT;
            const int py = (m / WOUT) % HOUT;
            const int ni = m / (WOUT * HOUT);
            if (OUTF32) {                            // f32 NCHW (final layer)
                if (nCh < COUT) {
                    const long o = ((ni * COUT + nCh) * (long)HOUT + py) * WOUT + px;
                    ((float*)outv)[o] = v;
                }
            } else {                                 // f16 NHWC
                const long o = ((long)(ni * HOUT + py) * WOUT + px) * COUTP + nCh;
                if (res) v += (float)res[o];
                ((_Float16*)outv)[o] = (_Float16)v;
            }
        }
    }
}

// Pack weights to f16 [CoutP][ky*kx][CinP]; handles OIHW (TRANS=0) and
// ConvTranspose (Cin,Cout,kh,kw) (TRANS=1); zero-pads Cin/Cout remainders.
__global__ __launch_bounds__(256)
void repack_weight_kernel(const float* __restrict__ w, _Float16* __restrict__ wp,
                          int COUT, int CIN, int KHW, int CINP, int TRANS, int total)
{
    const int gid = blockIdx.x * blockDim.x + threadIdx.x;
    if (gid >= total) return;
    const int ci   = gid % CINP;
    const int t    = gid / CINP;
    const int kofs = t % KHW;
    const int co   = t / KHW;
    float v = 0.0f;
    if (co < COUT && ci < CIN)
        v = TRANS ? w[(ci * COUT + co) * KHW + kofs]
                  : w[(co * CIN + ci) * KHW + kofs];
    wp[gid] = (_Float16)v;
}

template<int CIN, int CINP, int HIN, int WIN, int COUT, int COUTP,
         int HOUT, int WOUT, int KH, int KW, int S, int P, int TRANS,
         int RELU_IN, int ACT, int OUTF32, int NB>
static void launch_conv(const _Float16* in, const float* w, _Float16* wp,
                        const float* bias, const _Float16* res, void* out,
                        hipStream_t stream)
{
    constexpr int TOTAL = COUTP * KH * KW * CINP;
    constexpr int M     = 16 * HOUT * WOUT;          // batch = 16
    static_assert(CINP % 32 == 0 && COUTP % 16 == 0, "padding");
    static_assert((COUTP / 16) % NB == 0, "NB must divide N tiles");
    repack_weight_kernel<<<(TOTAL + 255) / 256, 256, 0, stream>>>(
        w, wp, COUT, CIN, KH * KW, CINP, TRANS, TOTAL);
    conv_wmma<CINP, HIN, WIN, COUT, COUTP, HOUT, WOUT, KH, KW, S, P, TRANS,
              RELU_IN, ACT, OUTF32, NB>
        <<<dim3(M / 16, (COUTP / 16) / NB, 1), dim3(32, 1, 1), 0, stream>>>(
            in, wp, bias, res, out);
}

// x (16,3,128,128) f32 NCHW -> (16,128,128,32) f16 NHWC, zero-padded channels
__global__ __launch_bounds__(256)
void nchw_to_nhwc_f16_kernel(const float* __restrict__ src, _Float16* __restrict__ dst)
{
    const int gid = blockIdx.x * blockDim.x + threadIdx.x;   // < 16*128*128*32
    const int ci  = gid & 31;
    const int p   = gid >> 5;               // n*16384 + y*128 + x
    const int yx  = p & 16383;
    const int n   = p >> 14;
    float v = 0.0f;
    if (ci < 3) v = src[(n * 3 + ci) * 16384 + yx];
    dst[gid] = (_Float16)v;
}

// ---------------------------------------------------------------------------
// Vector quantizer (D=256 dims, K=512 codes, HW=32*32 latent pixels, B=16)
// z is f16 NHWC: one pixel's 256-d row is contiguous.
// ---------------------------------------------------------------------------
__global__ __launch_bounds__(256)
void emb_norm_kernel(const float* __restrict__ emb, float* __restrict__ norms)
{
    const int c = blockIdx.x * blockDim.x + threadIdx.x;
    if (c < 512) {
        float s = 0.0f;
#pragma unroll 8
        for (int d = 0; d < 256; ++d) { const float e = emb[c * 256 + d]; s = fmaf(e, e, s); }
        norms[c] = s;
    }
}

// one wave per latent pixel; each lane scans 16 codes, cross-lane arg-min
__global__ __launch_bounds__(32)
void vq_argmin_kernel(const _Float16* __restrict__ z, const float* __restrict__ emb,
                      const float* __restrict__ norms, int* __restrict__ idx)
{
    constexpr int D = 256;
    const int row  = blockIdx.x;
    const int lane = threadIdx.x;
    const _Float16* __restrict__ zr = z + (long)row * D;

    float best  = 3.4e38f;
    int   bestc = 0;
    for (int g = 0; g < 16; ++g) {
        const int c = lane + g * 32;
        float dot = 0.0f;
#pragma unroll 4
        for (int d = 0; d < D; ++d)
            dot = fmaf((float)zr[d], emb[c * D + d], dot);
        const float score = norms[c] - 2.0f * dot;   // ||z||^2 constant for argmin
        if (score < best) { best = score; bestc = c; }
    }
    for (int off = 16; off > 0; off >>= 1) {
        const float ob = __shfl_xor(best, off, 32);
        const int   oc = __shfl_xor(bestc, off, 32);
        if (ob < best || (ob == best && oc < bestc)) { best = ob; bestc = oc; }
    }
    if (lane == 0) idx[row] = bestc;
}

__global__ __launch_bounds__(1)
void zero_scalar_kernel(float* p) { *p = 0.0f; }

// gather q = emb[idx] (f16 NHWC), accumulate sum((q-z)^2) via wave reduction
__global__ __launch_bounds__(256)
void vq_gather_loss_kernel(const _Float16* __restrict__ z, const float* __restrict__ emb,
                           const int* __restrict__ idx, _Float16* __restrict__ q,
                           float* __restrict__ lossAcc)
{
    const int gid = blockIdx.x * blockDim.x + threadIdx.x;   // < 16384*256
    const int row = gid >> 8;
    const int d   = gid & 255;
    const int c   = idx[row];
    const float qv = emb[c * 256 + d];
    const float dv = qv - (float)z[gid];
    q[gid] = (_Float16)qv;
    float sq = dv * dv;
    for (int off = 16; off > 0; off >>= 1) sq += __shfl_down(sq, off, 32);
    if ((threadIdx.x & 31) == 0) atomicAdd(lossAcc, sq);
}

__global__ __launch_bounds__(256)
void finalize_kernel(const int* __restrict__ idx, const float* __restrict__ lossAcc,
                     float* __restrict__ out)
{
    constexpr long xhatN = 16L * 3L * 128L * 128L;           // 786432
    const int i = blockIdx.x * blockDim.x + threadIdx.x;
    if (i < 16384) out[xhatN + 2 + i] = (float)idx[i];
    if (i == 0) {
        const float l = lossAcc[0] * (1.0f / (16384.0f * 256.0f));
        out[xhatN + 0] = l;   // dictionary_loss
        out[xhatN + 1] = l;   // commitment_loss
    }
}

// ---------------------------------------------------------------------------
extern "C" void kernel_launch(void* const* d_in, const int* in_sizes, int n_in,
                              void* d_out, int out_size, void* d_ws, size_t ws_size,
                              hipStream_t stream)
{
    (void)in_sizes; (void)n_in; (void)out_size; (void)ws_size;

    const float* x       = (const float*)d_in[0];
    const float* emb     = (const float*)d_in[1];
    const float* enc_w1  = (const float*)d_in[2];
    const float* enc_b1  = (const float*)d_in[3];
    const float* enc_w2  = (const float*)d_in[4];
    const float* enc_b2  = (const float*)d_in[5];
    const float* enc_w3  = (const float*)d_in[6];
    const float* enc_b3  = (const float*)d_in[7];
    const float* dec_w1  = (const float*)d_in[8];
    const float* dec_b1  = (const float*)d_in[9];
    const float* r1_w1   = (const float*)d_in[10];
    const float* r1_b1   = (const float*)d_in[11];
    const float* r1_w2   = (const float*)d_in[12];
    const float* r1_b2   = (const float*)d_in[13];
    const float* r2_w1   = (const float*)d_in[14];
    const float* r2_b1   = (const float*)d_in[15];
    const float* r2_w2   = (const float*)d_in[16];
    const float* r2_b2   = (const float*)d_in[17];
    const float* dect_w1 = (const float*)d_in[18];
    const float* dect_b1 = (const float*)d_in[19];
    const float* dect_w2 = (const float*)d_in[20];
    const float* dect_b2 = (const float*)d_in[21];

    float*     out = (float*)d_out;
    _Float16*  hws = (_Float16*)d_ws;

    // f16 NHWC activation regions (half offsets), reused along the timeline
    const long H_XH = 0;          // 16x128x128x32  (input, padded)
    const long H_Z1 = 8388608;    // 16x64x64x128
    const long H_Z2 = 16777216;   // 16x32x32x256
    const long H_Z3 = 20971520;   // 16x32x32x256
    const long H_Q  = 25165824;   // 16x32x32x256
    const long H_H  = 0;          // decoder trunk (reuses xh/z1 region)
    const long H_T  = 4194304;
    const long H_H2 = 8388608;
    const long H_H3 = 16777216;   // reuses z2 region
    const long H_U  = 0;          // 16x64x64x256

    const long H_MISC = 29360128;
    int*   idx     = (int*)(hws + H_MISC);           // 16384 ints (32768 halves)
    float* norms   = (float*)(hws + H_MISC + 32768); // 512 floats (1024 halves)
    float* lossAcc = (float*)(hws + H_MISC + 33792); // 1 float

    // packed f16 weights; 4,194,304 halves = 8 MB
    _Float16* wpB = hws + 29393936;
    _Float16* p_enc1  = wpB + 0;         // 128 x (16*32)
    _Float16* p_enc2  = wpB + 65536;     // 256 x (16*128)
    _Float16* p_enc3  = wpB + 589824;    // 256 x (9*256)
    _Float16* p_dec1  = wpB + 1179648;
    _Float16* p_r1w1  = wpB + 1769472;
    _Float16* p_r1w2  = wpB + 2359296;   // 256 x (1*256)
    _Float16* p_r2w1  = wpB + 2424832;
    _Float16* p_r2w2  = wpB + 3014656;
    _Float16* p_dect1 = wpB + 3080192;   // 256 x (16*256)
    _Float16* p_dect2 = wpB + 4128768;   // 16  x (16*256)

    // ---- input layout conversion ----
    nchw_to_nhwc_f16_kernel<<<32768, 256, 0, stream>>>(x, hws + H_XH);

    // ---- encoder ----
    //          CIN CINP HIN WIN COUT COUTP HOUT WOUT KH KW S  P  TR RI ACT F32 NB
    launch_conv<  3,  32,128,128, 128, 128,  64,  64, 4, 4, 2, 1, 0, 0, 1, 0, 8>(hws + H_XH, enc_w1, p_enc1, enc_b1, nullptr, hws + H_Z1, stream);
    launch_conv<128, 128, 64, 64, 256, 256,  32,  32, 4, 4, 2, 1, 0, 0, 1, 0, 8>(hws + H_Z1, enc_w2, p_enc2, enc_b2, nullptr, hws + H_Z2, stream);
    launch_conv<256, 256, 32, 32, 256, 256,  32,  32, 3, 3, 1, 1, 0, 0, 0, 0, 8>(hws + H_Z2, enc_w3, p_enc3, enc_b3, nullptr, hws + H_Z3, stream);

    // ---- vector quantizer ----
    emb_norm_kernel<<<2, 256, 0, stream>>>(emb, norms);
    vq_argmin_kernel<<<16384, 32, 0, stream>>>(hws + H_Z3, emb, norms, idx);
    zero_scalar_kernel<<<1, 1, 0, stream>>>(lossAcc);
    vq_gather_loss_kernel<<<16384, 256, 0, stream>>>(hws + H_Z3, emb, idx, hws + H_Q, lossAcc);

    // ---- decoder ----
    launch_conv<256, 256, 32, 32, 256, 256,  32,  32, 3, 3, 1, 1, 0, 0, 0, 0, 8>(hws + H_Q,  dec_w1, p_dec1, dec_b1, nullptr, hws + H_H,  stream);
    // residual 1
    launch_conv<256, 256, 32, 32, 256, 256,  32,  32, 3, 3, 1, 1, 0, 1, 0, 0, 8>(hws + H_H,  r1_w1, p_r1w1, r1_b1, nullptr,    hws + H_T,  stream);
    launch_conv<256, 256, 32, 32, 256, 256,  32,  32, 1, 1, 1, 0, 0, 1, 0, 0, 8>(hws + H_T,  r1_w2, p_r1w2, r1_b2, hws + H_H,  hws + H_H2, stream);
    // residual 2
    launch_conv<256, 256, 32, 32, 256, 256,  32,  32, 3, 3, 1, 1, 0, 1, 0, 0, 8>(hws + H_H2, r2_w1, p_r2w1, r2_b1, nullptr,    hws + H_T,  stream);
    launch_conv<256, 256, 32, 32, 256, 256,  32,  32, 1, 1, 1, 0, 0, 1, 0, 0, 8>(hws + H_T,  r2_w2, p_r2w2, r2_b2, hws + H_H2, hws + H_H3, stream);
    // transposed convs
    launch_conv<256, 256, 32, 32, 256, 256,  64,  64, 4, 4, 2, 1, 1, 0, 1, 0, 8>(hws + H_H3, dect_w1, p_dect1, dect_b1, nullptr, hws + H_U, stream);
    launch_conv<256, 256, 64, 64,   3,  16, 128, 128, 4, 4, 2, 1, 1, 0, 2, 1, 1>(hws + H_U,  dect_w2, p_dect2, dect_b2, nullptr, out,       stream);

    // ---- losses + indices ----
    finalize_kernel<<<64, 256, 0, stream>>>(idx, lossAcc, out);
}
